// ValueDecoder_20186346291821
// MI455X (gfx1250) — compile-verified, matrix operand order checked by reading, not run on hardware
//
#include <hip/hip_runtime.h>
#include <hip/hip_bf16.h>
#include <math.h>

// Problem constants (from reference)
#define BB     2048
#define SS     128
#define DD     1024
#define DFF    4096
#define NCAND  32
#define NTYPES 546

typedef __attribute__((ext_vector_type(16))) _Float16 v16h;
typedef __attribute__((ext_vector_type(8)))  _Float16 v8h;
typedef __attribute__((ext_vector_type(8)))  float    v8f;

// ---------------------------------------------------------------------------
// Kernel 1: per-batch argmax over candidates + gather selected encode row,
// convert to f16, and seed out[b] = b2.
// One block per batch; wave 0 (32 lanes == 32 candidates) does the ballot.
// ---------------------------------------------------------------------------
__global__ __launch_bounds__(256)
void vd_gather_rows(const int* __restrict__ cand,
                    const float* __restrict__ encode,
                    const float* __restrict__ b2,
                    _Float16* __restrict__ Xh,
                    float* __restrict__ out) {
    const int b = blockIdx.x;
    const int t = threadIdx.x;
    __shared__ int sfirst;

    if (t < 32) {  // exactly wave 0 on wave32 hardware
        const int c = cand[b * NCAND + t];
        const bool is_sent = (c == NTYPES) || (c == NTYPES + 1);
        unsigned mask;
#if __has_builtin(__builtin_amdgcn_ballot_w32)
        mask = __builtin_amdgcn_ballot_w32(is_sent);
#else
        mask = (unsigned)__ballot(is_sent);
#endif
        if (t == 0) {
            sfirst = mask ? (int)__builtin_ctz(mask) : 0;  // argmax semantics
            out[b] = b2[0];                                 // seed with bias2
        }
    }
    __syncthreads();

    const int s = SS - NCAND + sfirst;  // encode[:, -32:][first]
    const float* __restrict__ src = encode + ((size_t)b * SS + s) * DD;
    _Float16* __restrict__ dst = Xh + (size_t)b * DD;
    for (int i = t; i < DD; i += 256) dst[i] = (_Float16)src[i];
}

// ---------------------------------------------------------------------------
// Kernel 2: W1 (D x DFF, f32, row-major) -> W1t (DFF x D, f16) so the WMMA
// B-fragment reads K contiguously per output column. LDS-tiled transpose,
// both global accesses coalesced.
// ---------------------------------------------------------------------------
__global__ __launch_bounds__(256)
void vd_transpose_w1(const float* __restrict__ W1, _Float16* __restrict__ W1t) {
    __shared__ float tile[32][33];
    const int tx = threadIdx.x;            // 0..31
    const int ty = threadIdx.y;            // 0..7
    const int f0 = blockIdx.x * 32;        // along DFF
    const int d0 = blockIdx.y * 32;        // along D

#pragma unroll
    for (int i = 0; i < 4; ++i) {
        const int d = d0 + ty + i * 8;
        tile[ty + i * 8][tx] = W1[(size_t)d * DFF + f0 + tx];
    }
    __syncthreads();
#pragma unroll
    for (int i = 0; i < 4; ++i) {
        const int f = f0 + ty + i * 8;
        W1t[(size_t)f * DD + d0 + tx] = (_Float16)tile[tx][ty + i * 8];
    }
}

// ---------------------------------------------------------------------------
// Kernel 3: fused GEMM + bias + exact GeLU + dot(W2) reduction.
// Block = 256 threads = 8 waves. Block tile: M=16 batches x N=128 hidden.
// Each wave owns one 16x16 output tile; K=1024 -> 32 x v_wmma_f32_16x16x32_f16.
// A-panel staged in LDS (32 KB). Fragments follow ISA §7.12.2 16-bit layout:
//   lanes 0-15: K-chunks {kb+0..7, kb+16..23} with kb=0; lanes 16-31: kb=8.
// ---------------------------------------------------------------------------
__global__ __launch_bounds__(256)
void vd_gemm_gelu(const _Float16* __restrict__ Xh,
                  const _Float16* __restrict__ W1t,
                  const float* __restrict__ b1,
                  const float* __restrict__ W2,
                  float* __restrict__ out) {
    __shared__ _Float16 Xs[16 * DD];  // 32 KB

    const int b0   = blockIdx.x * 16;     // M block (batch rows)
    const int f0   = blockIdx.y * 128;    // N block (hidden cols)
    const int t    = threadIdx.x;
    const int wave = t >> 5;              // 0..7
    const int lane = t & 31;

    // Cooperative load of the 16 x 1024 f16 A panel (b128 transfers).
    {
        const v8h* __restrict__ srcv = (const v8h*)(Xh + (size_t)b0 * DD);
        v8h* __restrict__ dstv = (v8h*)Xs;
        for (int i = t; i < 16 * DD / 8; i += 256) dstv[i] = srcv[i];
    }
    __syncthreads();

    const int m   = lane & 15;               // A row within tile
    const int kb  = (lane >> 4) * 8;         // K sub-chunk base per half-wave
    const int col = f0 + wave * 16 + (lane & 15);  // B column (hidden index)
    const _Float16* __restrict__ Bp = W1t + (size_t)col * DD;

    v8f c = {};
    for (int k0 = 0; k0 < DD; k0 += 32) {
        // stream the B row through L2 (whole W1t fits in 192 MB L2 easily)
        __builtin_prefetch((const void*)(Bp + k0 + 128), 0, 1);

        const v8h alo = *(const v8h*)(Xs + m * DD + k0 + kb);
        const v8h ahi = *(const v8h*)(Xs + m * DD + k0 + kb + 16);
        const v8h blo = *(const v8h*)(Bp + k0 + kb);
        const v8h bhi = *(const v8h*)(Bp + k0 + kb + 16);

        v16h a, bmat;
#pragma unroll
        for (int i = 0; i < 8; ++i) {
            a[i] = alo[i];  a[i + 8] = ahi[i];
            bmat[i] = blo[i]; bmat[i + 8] = bhi[i];
        }
        // D(f32,16x16) = A(16x32 f16) * B(32x16 f16) + C
        c = __builtin_amdgcn_wmma_f32_16x16x32_f16(
                /*neg_a=*/false, a, /*neg_b=*/false, bmat,
                /*c_mod=*/(short)0, c, /*reuse_a=*/false, /*reuse_b=*/false);
    }

    // Epilogue: element (M = r + 8*(lane>=16), N = col).
    const float bias = b1[col];
    const float w2   = W2[col];
    float rowacc[8];
#pragma unroll
    for (int r = 0; r < 8; ++r) {
        const float v = c[r] + bias;
        const float g = 0.5f * v * (1.0f + erff(v * 0.70710678118654752440f));
        rowacc[r] = g * w2;  // contribution of this N to row-sum
    }
    // Reduce across the 16 N-lanes of each half-wave (same M set).
#pragma unroll
    for (int r = 0; r < 8; ++r) {
        float v = rowacc[r];
        v += __shfl_xor(v, 1);
        v += __shfl_xor(v, 2);
        v += __shfl_xor(v, 4);
        v += __shfl_xor(v, 8);
        rowacc[r] = v;
    }
    const int mbase = (lane >> 4) * 8;  // lane 0 -> rows 0..7, lane 16 -> rows 8..15
    if ((lane & 15) == 0) {
#pragma unroll
        for (int r = 0; r < 8; ++r)
            atomicAdd(&out[b0 + mbase + r], rowacc[r]);
    }
}

// ---------------------------------------------------------------------------
extern "C" void kernel_launch(void* const* d_in, const int* in_sizes, int n_in,
                              void* d_out, int out_size, void* d_ws, size_t ws_size,
                              hipStream_t stream) {
    (void)in_sizes; (void)n_in; (void)out_size; (void)ws_size;

    const int*   cand   = (const int*)d_in[0];
    const float* encode = (const float*)d_in[1];
    const float* W1     = (const float*)d_in[2];
    const float* b1     = (const float*)d_in[3];
    const float* W2     = (const float*)d_in[4];
    const float* b2     = (const float*)d_in[5];
    float* out = (float*)d_out;

    // Workspace layout: Xh (B*D f16 = 4 MB) | W1t (DFF*D f16 = 8 MB)
    _Float16* Xh  = (_Float16*)d_ws;
    _Float16* W1t = (_Float16*)((char*)d_ws + (size_t)BB * DD * sizeof(_Float16));

    // 1) argmax + gather + f16 convert + seed out with b2
    vd_gather_rows<<<BB, 256, 0, stream>>>(cand, encode, b2, Xh, out);

    // 2) W1 -> W1^T f16
    vd_transpose_w1<<<dim3(DFF / 32, DD / 32), dim3(32, 8), 0, stream>>>(W1, W1t);

    // 3) fused WMMA GEMM + bias + GeLU + dot(W2), atomic row reduction
    vd_gemm_gelu<<<dim3(BB / 16, DFF / 128), 256, 0, stream>>>(Xh, W1t, b1, W2, out);
}
